// MultiHeadMemory_26877905338696
// MI455X (gfx1250) — matrix operand order repaired and checked by reading, not compiled, not run on hardware
//
#include <hip/hip_runtime.h>
#include <hip/hip_bf16.h>

// Problem constants (match reference)
#define H_  16
#define M_  1024
#define D_  512
#define O_  512
#define N_  16384
#define HO_ (H_ * O_)

typedef __attribute__((ext_vector_type(16))) _Float16 v16h;
typedef __attribute__((ext_vector_type(8)))  _Float16 v8h;
typedef __attribute__((ext_vector_type(8)))  float    v8f;

// Matches the builtin's parameter type: vector_size(16) int
typedef int v4i_gcc __attribute__((vector_size(16)));
#define AS1 __attribute__((address_space(1)))
#define AS3 __attribute__((address_space(3)))

// Async global->LDS copy path (gfx1250). Guarded so the file compiles on
// toolchains that lack the builtins; fallback is a sync load + ds_store copy.
#if __has_builtin(__builtin_amdgcn_global_load_async_to_lds_b128) && \
    __has_builtin(__builtin_amdgcn_s_wait_asynccnt)
#define USE_ASYNC_LDS 1
#endif

static __device__ __forceinline__ v8f wmma_f16(v16h a, v16h b, v8f c) {
  // D = A(16x32 f16) * B(32x16 f16) + C(16x16 f32)
  return __builtin_amdgcn_wmma_f32_16x16x32_f16(
      /*neg_a=*/false, a, /*neg_b=*/false, b,
      /*c_mod=*/(short)0, c, /*reuse_a=*/false, /*reuse_b=*/false);
}

static __device__ __forceinline__ v8f vzero8() {
  v8f c;
#pragma unroll
  for (int i = 0; i < 8; ++i) c[i] = 0.0f;
  return c;
}

// A operand (16x32, MxK) from row-major f16, tile origin = base, row stride ld.
// ISA layout: lanes 0-15 row M=lane hold K 0-7 (v0-3) and 16-23 (v4-7);
// lanes 16-31 same rows hold K 8-15 and 24-31.
static __device__ __forceinline__ v16h load_a16(const _Float16* base, int ld, int lane) {
  const int row = lane & 15;
  const int hi  = (lane >> 4) << 3;  // 0 or 8
  const _Float16* p = base + (size_t)row * ld + hi;
  v8h x0 = *(const v8h*)(p);
  v8h x1 = *(const v8h*)(p + 16);
  v16h r;
#pragma unroll
  for (int i = 0; i < 8; ++i) { r[i] = x0[i]; r[i + 8] = x1[i]; }
  return r;
}

// B operand (32x16, KxN) where column j of B = row j of a row-major f16 matrix
// (i.e. computing X * Wrow^T). Lane j (0-15) holds K 0-7 and 16-23,
// lane j+16 holds K 8-15 and 24-31 of the same column.
static __device__ __forceinline__ v16h load_b16(const _Float16* base, int ld, int lane) {
  const int col = lane & 15;
  const int hi  = (lane >> 4) << 3;  // 0 or 8
  const _Float16* p = base + (size_t)col * ld + hi;
  v8h x0 = *(const v8h*)(p);
  v8h x1 = *(const v8h*)(p + 16);
  v16h r;
#pragma unroll
  for (int i = 0; i < 8; ++i) { r[i] = x0[i]; r[i + 8] = x1[i]; }
  return r;
}

// A operand from an LDS-staged 16x32 f32 tile (row-major, stride 32)
static __device__ __forceinline__ v16h load_a_lds(const float* st, int lane) {
  const int row = lane & 15;
  const int hi  = (lane >> 4) << 3;
  const float* p = st + row * 32 + hi;
  v16h r;
#pragma unroll
  for (int i = 0; i < 8; ++i) {
    r[i]     = (_Float16)p[i];
    r[i + 8] = (_Float16)p[i + 16];
  }
  return r;
}

// Reductions within aligned 16-lane groups (wave32: xor masks 1,2,4,8)
static __device__ __forceinline__ float groupMax16(float v) {
  v = fmaxf(v, __shfl_xor(v, 1, 32));
  v = fmaxf(v, __shfl_xor(v, 2, 32));
  v = fmaxf(v, __shfl_xor(v, 4, 32));
  v = fmaxf(v, __shfl_xor(v, 8, 32));
  return v;
}
static __device__ __forceinline__ float groupSum16(float v) {
  v += __shfl_xor(v, 1, 32);
  v += __shfl_xor(v, 2, 32);
  v += __shfl_xor(v, 4, 32);
  v += __shfl_xor(v, 8, 32);
  return v;
}

// ---------------------------------------------------------------------------
// Kernel 0: f32 -> f16 conversion
// ---------------------------------------------------------------------------
__global__ void cvt_f32_f16(const float* __restrict__ in, _Float16* __restrict__ out, size_t n) {
  size_t i = (size_t)blockIdx.x * blockDim.x + threadIdx.x;
  if (i < n) out[i] = (_Float16)in[i];
}

// ---------------------------------------------------------------------------
// Kernel 1: per head: mem_key = softmax(mems @ Wk^T + bk), mem_val^T
// Block: 16 rows of one head; 8 waves; wave w covers 64 output cols.
// ---------------------------------------------------------------------------
__global__ __launch_bounds__(256) void k1_keyval(
    const _Float16* __restrict__ mems, const _Float16* __restrict__ wk,
    const float* __restrict__ bk, const _Float16* __restrict__ wv,
    const float* __restrict__ bv,
    _Float16* __restrict__ memkey, _Float16* __restrict__ memvalT) {
  __shared__ float red[8 * 16];

  const int lane = threadIdx.x & 31;
  const int w    = threadIdx.x >> 5;
  const int m0   = blockIdx.x * 16;
  const int h    = blockIdx.y;
  const int o0   = w * 64;
  const int g    = lane >> 4;
  const int cl   = lane & 15;

  v8f ck[4], cv[4];
#pragma unroll
  for (int t = 0; t < 4; ++t) { ck[t] = vzero8(); cv[t] = vzero8(); }

  const _Float16* aBase = mems + ((size_t)h * M_ + m0) * D_;
  const _Float16* wkB   = wk + (size_t)h * O_ * D_;
  const _Float16* wvB   = wv + (size_t)h * O_ * D_;

  for (int d0 = 0; d0 < D_; d0 += 32) {
    v16h a = load_a16(aBase + d0, D_, lane);
#pragma unroll
    for (int t = 0; t < 4; ++t) {
      v16h bK = load_b16(wkB + (size_t)(o0 + 16 * t) * D_ + d0, D_, lane);
      ck[t] = wmma_f16(a, bK, ck[t]);
      v16h bV = load_b16(wvB + (size_t)(o0 + 16 * t) * D_ + d0, D_, lane);
      cv[t] = wmma_f16(a, bV, cv[t]);
    }
  }

  // bias add (column index constant per lane per tile)
#pragma unroll
  for (int t = 0; t < 4; ++t) {
    float bkv = bk[h * O_ + o0 + 16 * t + cl];
    float bvv = bv[h * O_ + o0 + 16 * t + cl];
#pragma unroll
    for (int r = 0; r < 8; ++r) { ck[t][r] += bkv; cv[t][r] += bvv; }
  }

  // exact softmax over O=512 (cross-wave via LDS)
  float rmax[8];
#pragma unroll
  for (int r = 0; r < 8; ++r) {
    float v = fmaxf(fmaxf(ck[0][r], ck[1][r]), fmaxf(ck[2][r], ck[3][r]));
    rmax[r] = groupMax16(v);
  }
  if (cl == 0) {
#pragma unroll
    for (int r = 0; r < 8; ++r) red[w * 16 + 8 * g + r] = rmax[r];
  }
  __syncthreads();
#pragma unroll
  for (int r = 0; r < 8; ++r) {
    float v = red[8 * g + r];
    for (int w2 = 1; w2 < 8; ++w2) v = fmaxf(v, red[w2 * 16 + 8 * g + r]);
    rmax[r] = v;
  }
  __syncthreads();

  float rsum[8];
#pragma unroll
  for (int r = 0; r < 8; ++r) {
    float s = 0.0f;
#pragma unroll
    for (int t = 0; t < 4; ++t) {
      float e = __expf(ck[t][r] - rmax[r]);
      ck[t][r] = e;
      s += e;
    }
    rsum[r] = groupSum16(s);
  }
  if (cl == 0) {
#pragma unroll
    for (int r = 0; r < 8; ++r) red[w * 16 + 8 * g + r] = rsum[r];
  }
  __syncthreads();
#pragma unroll
  for (int r = 0; r < 8; ++r) {
    float s = 0.0f;
    for (int w2 = 0; w2 < 8; ++w2) s += red[w2 * 16 + 8 * g + r];
    rsum[r] = 1.0f / s;
  }

  // store mem_key row-major, mem_val transposed [h][o][m]
#pragma unroll
  for (int t = 0; t < 4; ++t) {
    const int oc = o0 + 16 * t + cl;
#pragma unroll
    for (int r = 0; r < 8; ++r) {
      const int m = m0 + r + 8 * g;
      memkey[((size_t)h * M_ + m) * O_ + oc]  = (_Float16)(ck[t][r] * rsum[r]);
      memvalT[((size_t)h * O_ + oc) * M_ + m] = (_Float16)cv[t][r];
    }
  }
}

// ---------------------------------------------------------------------------
// Kernel 2: attention per head (streaming over M; no running-max needed since
// mem_key rows are convex weights => |logits| <= max|k| ~ 5, exp safe in f32).
// Block: 64 queries x 1 head; wave = 16 rows x 256 out cols.
// Q row-tile is held in registers; the 32x512 K-chunk is staged in LDS once
// per chunk (async global->LDS when available) and shared by all 8 waves.
// ---------------------------------------------------------------------------
#define KB_LD 536  // padded row stride (f16): 1072B = 16B aligned, bank-spread

__global__ __launch_bounds__(256) void k2_attn(
    const _Float16* __restrict__ kq, const _Float16* __restrict__ memkey,
    const _Float16* __restrict__ memvalT, _Float16* __restrict__ x) {
  __shared__ _Float16 kbuf[32 * KB_LD];   // ~33.5 KB staged K chunk
  __shared__ float stage[8][16 * 32];     // 16 KB wave-private P staging

  const int lane = threadIdx.x & 31;
  const int w    = threadIdx.x >> 5;
  const int rt   = w & 3;       // row tile 0..3
  const int ch   = w >> 2;      // col half 0..1
  const int n0   = blockIdx.x * 64 + rt * 16;
  const int h    = blockIdx.y;
  const int g    = lane >> 4;
  const int cl   = lane & 15;

  v8f acc[16];
#pragma unroll
  for (int t = 0; t < 16; ++t) acc[t] = vzero8();
  float lsum[8];
#pragma unroll
  for (int r = 0; r < 8; ++r) lsum[r] = 0.0f;

  const _Float16* qBase   = kq + (size_t)n0 * O_;
  const _Float16* keyBase = memkey + (size_t)h * M_ * O_;
  const _Float16* valBase = memvalT + ((size_t)h * O_ + ch * 256) * M_;
  float* st = stage[w];

  // Hoist the Q row tile (16x512 f16 = 16 x v16h = 128 VGPRs) into registers.
  v16h qa[16];
#pragma unroll
  for (int i = 0; i < 16; ++i) qa[i] = load_a16(qBase + i * 32, O_, lane);

  for (int m0 = 0; m0 < M_; m0 += 32) {
    // ---- Stage K chunk rows m0..m0+31 (32x512 f16) into LDS cooperatively.
    // 2048 x 16B segments / 256 threads = 8 per thread.
    {
      const int t = threadIdx.x;
#pragma unroll
      for (int i = 0; i < 8; ++i) {
        const int idx = i * 256 + t;   // 0..2047
        const int mm  = idx >> 6;      // row 0..31 (64 x 16B per 512-half row)
        const int seg = idx & 63;
        const _Float16* gp = keyBase + (size_t)(m0 + mm) * O_ + seg * 8;
        _Float16* lp = kbuf + mm * KB_LD + seg * 8;
#ifdef USE_ASYNC_LDS
        __builtin_amdgcn_global_load_async_to_lds_b128(
            (AS1 v4i_gcc*)gp, (AS3 v4i_gcc*)lp, 0, 0);
#else
        *(v8h*)lp = *(const v8h*)gp;
#endif
      }
#ifdef USE_ASYNC_LDS
      __builtin_amdgcn_s_wait_asynccnt(0);
#endif
    }
    __syncthreads();  // K chunk visible to all waves; prior stage reads done

    // ---- S chunk = Q(16x512) * K_chunk(32x512)^T -> 16x32 (B from LDS)
    v8f s0 = vzero8(), s1 = vzero8();
#pragma unroll
    for (int o = 0; o < 16; ++o) {
      v16h b0 = load_b16(kbuf + o * 32,               KB_LD, lane);
      v16h b1 = load_b16(kbuf + 16 * KB_LD + o * 32,  KB_LD, lane);
      s0 = wmma_f16(qa[o], b0, s0);
      s1 = wmma_f16(qa[o], b1, s1);
    }

    // ---- P = exp(S); accumulate row sums; stage P into A layout via LDS
#pragma unroll
    for (int r = 0; r < 8; ++r) {
      float e0 = __expf(s0[r]);
      float e1 = __expf(s1[r]);
      lsum[r] += groupSum16(e0 + e1);
      const int row = r + 8 * g;
      st[row * 32 + cl]      = e0;
      st[row * 32 + 16 + cl] = e1;
    }
    __syncthreads();
    v16h pa = load_a_lds(st, lane);
    __syncthreads();

    // ---- out += P(16x32) * V_chunk(32x256)
#pragma unroll
    for (int ct = 0; ct < 16; ++ct) {
      v16h vb = load_b16(valBase + (size_t)(ct * 16) * M_ + m0, M_, lane);
      acc[ct] = wmma_f16(pa, vb, acc[ct]);
    }
  }

#pragma unroll
  for (int r = 0; r < 8; ++r) lsum[r] = 1.0f / lsum[r];

#pragma unroll
  for (int ct = 0; ct < 16; ++ct) {
    const int oc = h * O_ + ch * 256 + ct * 16 + cl;
#pragma unroll
    for (int r = 0; r < 8; ++r) {
      const int n = n0 + r + 8 * g;
      x[(size_t)n * HO_ + oc] = (_Float16)(acc[ct][r] * lsum[r]);
    }
  }
}

// ---------------------------------------------------------------------------
// Kernel 3: final projection out = x[N, 8192] @ Wf^T + bf  (f32 output)
// Block: 64x64 tile; wave = 16 rows x 32 cols.
// ---------------------------------------------------------------------------
__global__ __launch_bounds__(256) void k3_proj(
    const _Float16* __restrict__ x, const _Float16* __restrict__ wf,
    const float* __restrict__ bf, float* __restrict__ out) {
  const int lane = threadIdx.x & 31;
  const int w    = threadIdx.x >> 5;
  const int rt   = w & 3;
  const int cw   = w >> 2;
  const int n0   = blockIdx.x * 64 + rt * 16;
  const int c0   = blockIdx.y * 64 + cw * 32;
  const int g    = lane >> 4;
  const int cl   = lane & 15;

  v8f acc0 = vzero8(), acc1 = vzero8();
  const _Float16* aBase  = x + (size_t)n0 * HO_;
  const _Float16* b0Base = wf + (size_t)c0 * HO_;
  const _Float16* b1Base = wf + (size_t)(c0 + 16) * HO_;

  for (int c = 0; c < HO_; c += 32) {
    v16h a  = load_a16(aBase + c, HO_, lane);
    v16h b0 = load_b16(b0Base + c, HO_, lane);
    v16h b1 = load_b16(b1Base + c, HO_, lane);
    acc0 = wmma_f16(a, b0, acc0);
    acc1 = wmma_f16(a, b1, acc1);
  }

  const float bf0 = bf[c0 + cl];
  const float bf1 = bf[c0 + 16 + cl];
#pragma unroll
  for (int r = 0; r < 8; ++r) {
    const int n = n0 + r + 8 * g;
    out[(size_t)n * O_ + c0 + cl]      = acc0[r] + bf0;
    out[(size_t)n * O_ + c0 + 16 + cl] = acc1[r] + bf1;
  }
}

// ---------------------------------------------------------------------------
extern "C" void kernel_launch(void* const* d_in, const int* in_sizes, int n_in,
                              void* d_out, int out_size, void* d_ws, size_t ws_size,
                              hipStream_t stream) {
  (void)in_sizes; (void)n_in; (void)out_size; (void)ws_size;

  const float* k    = (const float*)d_in[0];
  const float* mems = (const float*)d_in[1];
  const float* Wk   = (const float*)d_in[2];
  const float* bk   = (const float*)d_in[3];
  const float* Wv   = (const float*)d_in[4];
  const float* bv   = (const float*)d_in[5];
  const float* Wf   = (const float*)d_in[6];
  const float* bf   = (const float*)d_in[7];
  float* outp       = (float*)d_out;

  // Workspace layout (f16 elements)
  _Float16* p = (_Float16*)d_ws;
  _Float16* kf16    = p; p += (size_t)N_ * O_;        // 8.4M
  _Float16* memsf16 = p; p += (size_t)H_ * M_ * D_;   // 8.4M
  _Float16* wkf16   = p; p += (size_t)H_ * O_ * D_;   // 4.2M
  _Float16* wvf16   = p; p += (size_t)H_ * O_ * D_;   // 4.2M
  _Float16* wff16   = p; p += (size_t)O_ * HO_;       // 4.2M
  _Float16* memkey  = p; p += (size_t)H_ * M_ * O_;   // 8.4M
  _Float16* memvalT = p; p += (size_t)H_ * M_ * O_;   // 8.4M
  _Float16* xcat    = p; p += (size_t)N_ * HO_;       // 134.2M

  auto launch_cvt = [&](const float* src, _Float16* dst, size_t n) {
    int blocks = (int)((n + 255) / 256);
    hipLaunchKernelGGL(cvt_f32_f16, dim3(blocks), dim3(256), 0, stream, src, dst, n);
  };
  launch_cvt(k,    kf16,    (size_t)N_ * O_);
  launch_cvt(mems, memsf16, (size_t)H_ * M_ * D_);
  launch_cvt(Wk,   wkf16,   (size_t)H_ * O_ * D_);
  launch_cvt(Wv,   wvf16,   (size_t)H_ * O_ * D_);
  launch_cvt(Wf,   wff16,   (size_t)O_ * HO_);

  hipLaunchKernelGGL(k1_keyval, dim3(M_ / 16, H_), dim3(256), 0, stream,
                     memsf16, wkf16, bk, wvf16, bv, memkey, memvalT);

  hipLaunchKernelGGL(k2_attn, dim3(N_ / 64, H_), dim3(256), 0, stream,
                     kf16, memkey, memvalT, xcat);

  hipLaunchKernelGGL(k3_proj, dim3(N_ / 64, O_ / 64), dim3(256), 0, stream,
                     xcat, wff16, bf, outp);
}
